// GATRegressor_27582279975364
// MI455X (gfx1250) — matrix operand order repaired and checked by reading, not compiled, not run on hardware
//
#include <hip/hip_runtime.h>
#include <hip/hip_bf16.h>
#include <math.h>

typedef __attribute__((ext_vector_type(16))) _Float16 v16h;
typedef __attribute__((ext_vector_type(8)))  _Float16 v8h;
typedef __attribute__((ext_vector_type(8)))  float    v8f;

#define HEADS 4
#define HID   32
#define HF    128
#define NEG_SLOPE 0.2f

static __device__ __forceinline__ float elu1(float x) {
  return x > 0.f ? x : (expf(x) - 1.f);
}

// ---------------------------------------------------------------------------
// W [K,128] f32 row-major  ->  Wt [128][K] f16 (per-output-column contiguous)
// ---------------------------------------------------------------------------
__global__ void convert_w_kernel(const float* __restrict__ W,
                                 _Float16* __restrict__ Wt, int K)
{
  int i = blockIdx.x * blockDim.x + threadIdx.x;
  if (i >= K * HF) return;
  int k = i / HF, c = i - k * HF;
  Wt[c * K + k] = (_Float16)W[i];
}

// flat f32 -> f16 convert (for the input features x)
__global__ void convert_x_kernel(const float* __restrict__ X,
                                 _Float16* __restrict__ Xh, long long n)
{
  long long i = (long long)blockIdx.x * blockDim.x + threadIdx.x;
  if (i < n) Xh[i] = (_Float16)X[i];
}

// ---------------------------------------------------------------------------
// Y[N,128] = X[N,K] @ W[K,128]  (f16 WMMA, f32 accumulate); X already f16.
// One wave -> one 16x16 tile of Y. Block = 256 thr = 8 waves = 16 rows x 128 cols.
// X tile staged via GLOBAL_LOAD_ASYNC_TO_LDS_B128 (ASYNCcnt), bank-padded;
// A fragment = two ds_load_b128 per k-chunk, no in-loop conversion;
// B fragment = one 32B contiguous f16 vector from transposed weights.
// ---------------------------------------------------------------------------
template<int K>
__global__ __launch_bounds__(256)
void linear_wmma_kernel(const _Float16* __restrict__ X,
                        const _Float16* __restrict__ Wt,
                        float* __restrict__ Y, int N)
{
  constexpr int KP = K + 8;               // padded LDS row stride (halfs, 16B mult)
  __shared__ _Float16 Xs[16 * KP];
  const int tid     = threadIdx.x;
  const int wave    = tid >> 5;
  const int lane    = tid & 31;
  const int rowBase = blockIdx.x * 16;
  const bool fullTile = (rowBase + 16 <= N);   // block-uniform

  if (fullTile) {
    constexpr int CPR    = K / 8;          // 16B chunks per row
    constexpr int CHUNKS = 16 * CPR;       // total 16B chunks (128 or 256)
    const unsigned ldsBase = (unsigned)(unsigned long long)(&Xs[0]);
    if (tid < CHUNKS) {                    // wave-uniform predicate
      int r = tid / CPR;
      int c = (tid - r * CPR) * 8;         // half index within row
      const _Float16* g = X + (rowBase + r) * K + c;
      unsigned lds = ldsBase + (unsigned)(r * KP + c) * 2u;
      asm volatile("global_load_async_to_lds_b128 %0, %1, off"
                   :: "v"(lds), "v"(g) : "memory");
    }
    asm volatile("s_wait_asynccnt 0" ::: "memory");
  } else {
    for (int i = tid; i < 16 * K; i += 256) {
      int r = i / K, c = i - r * K;
      int row = rowBase + r; if (row >= N) row = N - 1;
      Xs[r * KP + c] = X[row * K + c];
    }
  }
  __syncthreads();

  const int nIdx = lane & 15;             // row (A) / col (B,D) within tile
  const int half = lane >> 4;             // K sub-block select
  const int col  = wave * 16 + nIdx;
  const _Float16* arow = &Xs[nIdx * KP];

  v8f acc = {};
#pragma unroll
  for (int kk = 0; kk < K; kk += 32) {
    // A 16x32 f16 layout: lanes 0-15 (row=lane) hold K kk..kk+15,
    // lanes 16-31 hold K kk+16..kk+31.
    const int kbA = kk + half * 16;
    v8h lo = *(const v8h*)(arow + kbA);
    v8h hi = *(const v8h*)(arow + kbA + 8);
    v16h a = __builtin_shufflevector(lo, hi, 0, 1, 2, 3, 4, 5, 6, 7,
                                     8, 9, 10, 11, 12, 13, 14, 15);

    // B 32x16 f16 layout: lanes 0-15 (col=lane) hold K kk..kk+15,
    // lanes 16-31 hold K kk+16..kk+31. Contiguous 32B in transposed Wt.
    v16h b = *(const v16h*)(Wt + col * K + kk + half * 16);

    acc = __builtin_amdgcn_wmma_f32_16x16x32_f16(
        /*neg_a=*/false, a, /*neg_b=*/false, b,
        /*c_mod=*/(short)0, acc, /*reuse_a=*/false, /*reuse_b=*/false);
  }

  // D layout: VGPR r = row r (lanes 0-15) / row r+8 (lanes 16-31), col = lane&15
  float* yp = Y + (rowBase + half * 8) * HF + col;
  if (fullTile) {
#pragma unroll
    for (int r = 0; r < 8; ++r) yp[r * HF] = acc[r];
  } else {
#pragma unroll
    for (int r = 0; r < 8; ++r)
      if (rowBase + half * 8 + r < N) yp[r * HF] = acc[r];
  }
}

// ---------------------------------------------------------------------------
// per-(node,head) attention terms:  as_n = <h, a_src>, ad_n = <h, a_dst>
// ---------------------------------------------------------------------------
__global__ void attn_terms_kernel(const float* __restrict__ H,
                                  const float* __restrict__ a_src,
                                  const float* __restrict__ a_dst,
                                  float* __restrict__ asn,
                                  float* __restrict__ adn, int N)
{
  int idx = blockIdx.x * blockDim.x + threadIdx.x;
  if (idx >= N * HEADS) return;
  int n = idx >> 2, h = idx & 3;
  const float* hp = H + (long long)n * HF + h * HID;
  const float* sp = a_src + h * HID;
  const float* dp = a_dst + h * HID;
  float s = 0.f, d = 0.f;
#pragma unroll
  for (int f = 0; f < HID; ++f) { float v = hp[f]; s += v * sp[f]; d += v * dp[f]; }
  asn[idx] = s; adn[idx] = d;
}

__global__ void fill_kernel(float* __restrict__ p, long long n, float v) {
  long long i = (long long)blockIdx.x * blockDim.x + threadIdx.x;
  if (i < n) p[i] = v;
}

static __device__ __forceinline__ void edge_src_dst(const int* __restrict__ ei,
                                                    int ne, long long e,
                                                    int& s, int& d) {
  if (e < ne) { s = ei[e]; d = ei[(long long)ne + e]; }
  else        { s = d = (int)(e - ne); }            // appended self-loops
}

static __device__ __forceinline__ void atomicMaxF32(float* addr, float val) {
  unsigned int* ai = (unsigned int*)addr;
  unsigned int old = *ai;
  while (true) {
    float cur = __uint_as_float(old);
    if (cur >= val) break;
    unsigned int prev = atomicCAS(ai, old, __float_as_uint(val));
    if (prev == old) break;
    old = prev;
  }
}

// ---------------------------------------------------------------------------
// segment-softmax passes over edges (thread per (edge, head))
// ---------------------------------------------------------------------------
__global__ void edge_max_kernel(const int* __restrict__ ei, int ne, int nn,
                                const float* __restrict__ asn,
                                const float* __restrict__ adn,
                                float* __restrict__ mmax)
{
  long long idx = (long long)blockIdx.x * blockDim.x + threadIdx.x;
  long long tot = ((long long)ne + nn) * HEADS;
  if (idx >= tot) return;
  long long e = idx >> 2; int h = (int)(idx & 3);
  int s, d; edge_src_dst(ei, ne, e, s, d);
  float l = asn[(long long)s * HEADS + h] + adn[(long long)d * HEADS + h];
  l = l > 0.f ? l : NEG_SLOPE * l;
  atomicMaxF32(&mmax[(long long)d * HEADS + h], l);
}

__global__ void edge_expsum_kernel(const int* __restrict__ ei, int ne, int nn,
                                   const float* __restrict__ asn,
                                   const float* __restrict__ adn,
                                   const float* __restrict__ mmax,
                                   float* __restrict__ denom,
                                   float* __restrict__ alpha)
{
  long long idx = (long long)blockIdx.x * blockDim.x + threadIdx.x;
  long long tot = ((long long)ne + nn) * HEADS;
  if (idx >= tot) return;
  long long e = idx >> 2; int h = (int)(idx & 3);
  int s, d; edge_src_dst(ei, ne, e, s, d);
  float l = asn[(long long)s * HEADS + h] + adn[(long long)d * HEADS + h];
  l = l > 0.f ? l : NEG_SLOPE * l;
  float ev = expf(l - mmax[(long long)d * HEADS + h]);
  alpha[idx] = ev;
  atomicAdd(&denom[(long long)d * HEADS + h], ev);
}

__global__ void edge_scatter_kernel(const int* __restrict__ ei, int ne, int nn,
                                    const float* __restrict__ alpha,
                                    const float* __restrict__ denom,
                                    const float* __restrict__ Hlin,
                                    float* __restrict__ Hagg)
{
  long long idx = (long long)blockIdx.x * blockDim.x + threadIdx.x;
  long long tot = ((long long)ne + nn) * HEADS;
  if (idx >= tot) return;
  long long e = idx >> 2; int h = (int)(idx & 3);
  int s, d; edge_src_dst(ei, ne, e, s, d);
  float w = alpha[idx] / denom[(long long)d * HEADS + h];
  const float* src = Hlin + (long long)s * HF + h * HID;
  float*       dst = Hagg + (long long)d * HF + h * HID;
#pragma unroll
  for (int f = 0; f < HID; f += 4) {
    float4 v = *(const float4*)(src + f);
    atomicAdd(dst + f + 0, v.x * w);
    atomicAdd(dst + f + 1, v.y * w);
    atomicAdd(dst + f + 2, v.z * w);
    atomicAdd(dst + f + 3, v.w * w);
  }
}

// ---------------------------------------------------------------------------
// bias + ELU, emitting f16 activations ready for the next WMMA layer
// ---------------------------------------------------------------------------
__global__ void finalize_elu_kernel(const float* __restrict__ Hagg,
                                    const float* __restrict__ bias,
                                    _Float16* __restrict__ Hout, int N)
{
  long long i = (long long)blockIdx.x * blockDim.x + threadIdx.x;
  if (i >= (long long)N * HF) return;
  int c = (int)(i & (HF - 1));
  Hout[i] = (_Float16)elu1(Hagg[i] + bias[c]);
}

__global__ void finalize_mean_elu_kernel(const float* __restrict__ Hagg,
                                         const float* __restrict__ bias,
                                         float* __restrict__ Hout, int N)
{
  long long i = (long long)blockIdx.x * blockDim.x + threadIdx.x;
  if (i >= (long long)N * HID) return;
  long long n = i / HID; int f = (int)(i - n * HID);
  const float* hp = Hagg + n * HF;
  float m = 0.25f * (hp[f] + hp[HID + f] + hp[2 * HID + f] + hp[3 * HID + f]);
  Hout[i] = elu1(m + bias[f]);
}

__global__ void pool_kernel(const float* __restrict__ H3,
                            const int* __restrict__ batch,
                            float* __restrict__ gsum,
                            float* __restrict__ gcnt, int N)
{
  long long i = (long long)blockIdx.x * blockDim.x + threadIdx.x;
  if (i >= (long long)N * HID) return;
  long long n = i / HID; int f = (int)(i - n * HID);
  int g = batch[n];
  atomicAdd(&gsum[(long long)g * HID + f], H3[i]);
  if (f == 0) atomicAdd(&gcnt[g], 1.f);
}

__global__ void mlp_kernel(const float* __restrict__ gsum,
                           const float* __restrict__ gcnt,
                           const float* __restrict__ Wm1,
                           const float* __restrict__ bm1,
                           const float* __restrict__ Wm2,
                           const float* __restrict__ bm2,
                           float* __restrict__ out, int G)
{
  int g = blockIdx.x * blockDim.x + threadIdx.x;
  if (g >= G) return;
  float inv = 1.f / fmaxf(gcnt[g], 1.f);
  float pooled[HID];
#pragma unroll
  for (int f = 0; f < HID; ++f) pooled[f] = gsum[(long long)g * HID + f] * inv;
  float acc = bm2[0];
  for (int j = 0; j < 64; ++j) {
    float z = bm1[j];
#pragma unroll
    for (int f = 0; f < HID; ++f) z += pooled[f] * Wm1[f * 64 + j];
    z = fmaxf(z, 0.f);
    acc += z * Wm2[j];
  }
  out[g] = acc;
}

// ---------------------------------------------------------------------------
extern "C" void kernel_launch(void* const* d_in, const int* in_sizes, int n_in,
                              void* d_out, int out_size, void* d_ws, size_t ws_size,
                              hipStream_t stream)
{
  (void)n_in; (void)ws_size;
  const float* x     = (const float*)d_in[0];
  const int*   ei    = (const int*)d_in[1];
  const int*   batch = (const int*)d_in[2];
  const float* W[3]  = {(const float*)d_in[3], (const float*)d_in[7],  (const float*)d_in[11]};
  const float* aS[3] = {(const float*)d_in[4], (const float*)d_in[8],  (const float*)d_in[12]};
  const float* aD[3] = {(const float*)d_in[5], (const float*)d_in[9],  (const float*)d_in[13]};
  const float* bb[3] = {(const float*)d_in[6], (const float*)d_in[10], (const float*)d_in[14]};
  const float* Wm1 = (const float*)d_in[15];
  const float* bm1 = (const float*)d_in[16];
  const float* Wm2 = (const float*)d_in[17];
  const float* bm2 = (const float*)d_in[18];

  const int N  = in_sizes[0] / 64;        // IN_DIM = 64
  const int NE = in_sizes[1] / 2;         // edge_index is [2, NE]
  const int G  = out_size;
  const long long Et = (long long)NE + N; // + self loops

  char* p = (char*)d_ws;
  auto carveb = [&](size_t bytes) -> char* {
    char* r = p;
    p += (bytes + 255) & ~(size_t)255;
    return r;
  };
  float*    hlin  = (float*)carveb((size_t)N * HF * 4);
  float*    hagg  = (float*)carveb((size_t)N * HF * 4);
  _Float16* hcur  = (_Float16*)carveb((size_t)N * HF * 2);
  _Float16* xh    = (_Float16*)carveb((size_t)N * 64 * 2);
  float*    asn   = (float*)carveb((size_t)N * HEADS * 4);
  float*    adn   = (float*)carveb((size_t)N * HEADS * 4);
  float*    mmax  = (float*)carveb((size_t)N * HEADS * 4);
  float*    denom = (float*)carveb((size_t)N * HEADS * 4);
  float*    alpha = (float*)carveb((size_t)Et * HEADS * 4);
  float*    h3    = (float*)carveb((size_t)N * HID * 4);
  float*    gsum  = (float*)carveb((size_t)G * HID * 4);
  float*    gcnt  = (float*)carveb((size_t)G * 4);
  _Float16* wt[3];
  wt[0] = (_Float16*)carveb((size_t)64  * HF * 2);
  wt[1] = (_Float16*)carveb((size_t)128 * HF * 2);
  wt[2] = (_Float16*)carveb((size_t)128 * HF * 2);

  const int B = 256;
  const int gLin  = (N + 15) / 16;
  const int gNH   = (int)(((long long)N * HEADS + B - 1) / B);
  const int gNHF  = (int)(((long long)N * HF + B - 1) / B);
  const int gN32  = (int)(((long long)N * HID + B - 1) / B);
  const int gEdge = (int)((Et * HEADS + B - 1) / B);

  // one-time precision conversions (f32 -> f16)
  convert_w_kernel<<<(64  * HF + B - 1) / B, B, 0, stream>>>(W[0], wt[0], 64);
  convert_w_kernel<<<(128 * HF + B - 1) / B, B, 0, stream>>>(W[1], wt[1], 128);
  convert_w_kernel<<<(128 * HF + B - 1) / B, B, 0, stream>>>(W[2], wt[2], 128);
  convert_x_kernel<<<(int)(((long long)N * 64 + B - 1) / B), B, 0, stream>>>(
      x, xh, (long long)N * 64);

  for (int L = 0; L < 3; ++L) {
    if (L == 0)
      linear_wmma_kernel<64><<<gLin, B, 0, stream>>>(xh, wt[0], hlin, N);
    else
      linear_wmma_kernel<128><<<gLin, B, 0, stream>>>(hcur, wt[L], hlin, N);

    attn_terms_kernel<<<gNH, B, 0, stream>>>(hlin, aS[L], aD[L], asn, adn, N);
    fill_kernel<<<gNH, B, 0, stream>>>(mmax,  (long long)N * HEADS, -3.0e38f);
    fill_kernel<<<gNH, B, 0, stream>>>(denom, (long long)N * HEADS, 0.f);
    fill_kernel<<<gNHF, B, 0, stream>>>(hagg, (long long)N * HF, 0.f);

    edge_max_kernel<<<gEdge, B, 0, stream>>>(ei, NE, N, asn, adn, mmax);
    edge_expsum_kernel<<<gEdge, B, 0, stream>>>(ei, NE, N, asn, adn, mmax, denom, alpha);
    edge_scatter_kernel<<<gEdge, B, 0, stream>>>(ei, NE, N, alpha, denom, hlin, hagg);

    if (L < 2)
      finalize_elu_kernel<<<gNHF, B, 0, stream>>>(hagg, bb[L], hcur, N);
    else
      finalize_mean_elu_kernel<<<gN32, B, 0, stream>>>(hagg, bb[2], h3, N);
  }

  fill_kernel<<<(int)(((long long)G * HID + B - 1) / B), B, 0, stream>>>(gsum, (long long)G * HID, 0.f);
  fill_kernel<<<(G + B - 1) / B, B, 0, stream>>>(gcnt, G, 0.f);
  pool_kernel<<<gN32, B, 0, stream>>>(h3, batch, gsum, gcnt, N);
  mlp_kernel<<<(G + B - 1) / B, B, 0, stream>>>(gsum, gcnt, Wm1, bm1, Wm2, bm2,
                                                (float*)d_out, G);
}
